// SoftRAMAttention_85950885527684
// MI455X (gfx1250) — matrix operand order repaired and checked by reading, not compiled, not run on hardware
//
#include <hip/hip_runtime.h>
#include <hip/hip_bf16.h>

// SoftRAMAttention for MI455X (gfx1250, wave32).
// S=2048, B=64, H=8, K=12, N_POS=11.
//
// Phase 1: build weight matrices so address formation becomes GEMMs,
//          and bit-pack head_table (0/1 values) into 8*4096 bits.
// Phase 2: WMMA (v_wmma_f32_16x16x32_f16) GEMMs computing
//          addr_q/addr_k (tokens @ Wqk), addr_p (posb @ Wp), addr_v (tokens @ Wv).
//          Weights split into lo=w&63 / hi=w>>6 (f16-exact); addr = 64*hi+lo.
// Phase 3: causal vote scan per row (bit lookups in 4KB LDS table),
//          first-argmax via monotone key, then table_v gather for output.

#define SDIM 2048
#define BDIM 64
#define HDIM 8
#define KDIM 12
#define NPOS 11

typedef __attribute__((ext_vector_type(16))) _Float16 v16h;
typedef __attribute__((ext_vector_type(8)))  float    v8f;

// ---------------------------------------------------------------------------
// Kernel 1: build weights + bit-packed head table.
// ---------------------------------------------------------------------------
__global__ void softram_build(const int* __restrict__ conn_heads,   // [8][12]
                              const float* __restrict__ head_table, // [8][4096]
                              const int* __restrict__ conn_v,       // [64][12]
                              _Float16* __restrict__ Wqk_lo,        // [64][16]
                              _Float16* __restrict__ Wqk_hi,
                              _Float16* __restrict__ Wp_lo,         // [32][16]
                              _Float16* __restrict__ Wp_hi,
                              _Float16* __restrict__ Wv_lo,         // [64][64]
                              _Float16* __restrict__ Wv_hi,
                              unsigned int* __restrict__ packedT)   // [8][128]
{
    const int tid = threadIdx.x;

    // Wqk: rows b'=0..63 (token bit), cols 0..7 = query weights, 8..15 = key weights
    for (int idx = tid; idx < 64 * 16; idx += 256) {
        int b = idx >> 4, h = idx & 15;
        int head   = (h < 8) ? h : (h - 8);
        int target = (h < 8) ? b : (64 + b);
        int w = 0;
        #pragma unroll
        for (int k = 0; k < KDIM; ++k)
            if (conn_heads[head * KDIM + k] == target) w += (1 << (KDIM - 1 - k));
        Wqk_lo[idx] = (_Float16)(float)(w & 63);
        Wqk_hi[idx] = (_Float16)(float)(w >> 6);
    }
    // Wp: rows p=0..31 (only 0..10 used), cols 0..7 used
    for (int idx = tid; idx < 32 * 16; idx += 256) {
        int p = idx >> 4, h = idx & 15;
        int w = 0;
        if (p < NPOS && h < 8) {
            #pragma unroll
            for (int k = 0; k < KDIM; ++k)
                if (conn_heads[h * KDIM + k] == 128 + p) w += (1 << (KDIM - 1 - k));
        }
        Wp_lo[idx] = (_Float16)(float)(w & 63);
        Wp_hi[idx] = (_Float16)(float)(w >> 6);
    }
    // Wv: rows b'=0..63 (token bit), cols b=0..63 (value slot)
    for (int idx = tid; idx < 64 * 64; idx += 256) {
        int bp = idx >> 6, b = idx & 63;
        int w = 0;
        #pragma unroll
        for (int k = 0; k < KDIM; ++k)
            if (conn_v[b * KDIM + k] == bp) w += (1 << (KDIM - 1 - k));
        Wv_lo[idx] = (_Float16)(float)(w & 63);
        Wv_hi[idx] = (_Float16)(float)(w >> 6);
    }
    // Bit-pack head_table (entries are exactly 0.0 or 1.0)
    for (int idx = tid; idx < 8 * 128; idx += 256) {
        int h = idx >> 7, wd = idx & 127;
        unsigned int bits = 0;
        #pragma unroll
        for (int t = 0; t < 32; ++t)
            if (head_table[h * 4096 + wd * 32 + t] > 0.5f) bits |= (1u << t);
        packedT[idx] = bits;
    }
}

// ---------------------------------------------------------------------------
// WMMA helpers (16x16x32 f16 -> f32). wave32, one wave per block.
// A layout (16-bit A 16x32): lane<16: row=lane, halfs 0-7 = K(k0..k0+7),
//   halfs 8-15 = K(k0+16..k0+23); lane>=16: row=lane-16, K offsets +8 / +24.
// B layout (32x16): lane<16: col=lane, halfs 0-7 = B[k0..k0+7][col], 8-15 = B[k0+16..k0+23][col];
//   lane>=16: col=lane-16, K offsets +8 / +24 (same k order as A).
// C/D layout: lane<16: N=lane, vgpr r -> M=r; lane>=16: N=lane-16, M=r+8.
// ---------------------------------------------------------------------------
__device__ inline v16h load_A_tokens(const int* __restrict__ tok, int m0, int k0, int lane)
{
    int row = m0 + (lane & 15);
    int kb  = k0 + ((lane >> 4) << 3);   // +8 for lanes 16..31
    v16h a;
    #pragma unroll
    for (int t = 0; t < 8; ++t) a[t]     = (_Float16)(float)tok[row * 64 + kb + t];
    #pragma unroll
    for (int t = 0; t < 8; ++t) a[8 + t] = (_Float16)(float)tok[row * 64 + kb + 16 + t];
    return a;
}

__device__ inline v16h load_A_posb(int m0, int lane)
{
    int d  = m0 + (lane & 15);
    int kb = (lane >> 4) << 3;
    v16h a;
    #pragma unroll
    for (int t = 0; t < 8; ++t) {
        int p = kb + t;
        a[t] = (_Float16)((p < NPOS) ? (float)((d >> (NPOS - 1 - p)) & 1) : 0.0f);
    }
    #pragma unroll
    for (int t = 0; t < 8; ++t) {
        int p = kb + 16 + t;   // always >= 16 > 10 -> zero
        a[8 + t] = (_Float16)((p < NPOS) ? (float)((d >> (NPOS - 1 - p)) & 1) : 0.0f);
    }
    return a;
}

__device__ inline v16h load_B(const _Float16* __restrict__ W, int k0, int n0, int ncols, int lane)
{
    int col = n0 + (lane & 15);
    int kb  = k0 + ((lane >> 4) << 3);   // +8 for lanes 16..31
    v16h b;
    #pragma unroll
    for (int t = 0; t < 16; ++t) b[t] = W[(kb + (t & 7) + ((t >> 3) << 4)) * ncols + col];
    return b;
}

#define WMMA_F16(A, B, C) \
    __builtin_amdgcn_wmma_f32_16x16x32_f16(false, (A), false, (B), (short)0, (C), false, false)

// ---------------------------------------------------------------------------
// Kernel 2: address GEMMs via WMMA. One wave (32 threads) per 16-row tile.
// ---------------------------------------------------------------------------
__global__ void softram_addr(const int* __restrict__ tokens,
                             const _Float16* __restrict__ Wqk_lo,
                             const _Float16* __restrict__ Wqk_hi,
                             const _Float16* __restrict__ Wp_lo,
                             const _Float16* __restrict__ Wp_hi,
                             const _Float16* __restrict__ Wv_lo,
                             const _Float16* __restrict__ Wv_hi,
                             unsigned short* __restrict__ aq,   // [2048][8]
                             unsigned short* __restrict__ ak,   // [2048][8]
                             unsigned short* __restrict__ ap,   // [2048][8]
                             unsigned short* __restrict__ av)   // [2048][64]
{
    const int lane = threadIdx.x;
    const int m0   = blockIdx.x * 16;
    const int n    = lane & 15;
    const int mb   = (lane >> 4) << 3;

    // ---- addr_q / addr_k (tokens @ Wqk), cols 0-7 = q, 8-15 = k ----
    {
        v8f lo = {}; v8f hi = {};
        #pragma unroll
        for (int k0 = 0; k0 < 64; k0 += 32) {
            v16h a  = load_A_tokens(tokens, m0, k0, lane);
            v16h bl = load_B(Wqk_lo, k0, 0, 16, lane);
            v16h bh = load_B(Wqk_hi, k0, 0, 16, lane);
            lo = WMMA_F16(a, bl, lo);
            hi = WMMA_F16(a, bh, hi);
        }
        #pragma unroll
        for (int r = 0; r < 8; ++r) {
            int val = (int)(lo[r] + 64.0f * hi[r] + 0.5f);
            int M = m0 + mb + r;
            if (n < 8) aq[M * 8 + n] = (unsigned short)val;
            else       ak[M * 8 + (n - 8)] = (unsigned short)val;
        }
    }

    // ---- addr_p (posb @ Wp) ----
    {
        v8f lo = {}; v8f hi = {};
        v16h a = load_A_posb(m0, lane);
        lo = WMMA_F16(a, load_B(Wp_lo, 0, 0, 16, lane), lo);
        hi = WMMA_F16(a, load_B(Wp_hi, 0, 0, 16, lane), hi);
        #pragma unroll
        for (int r = 0; r < 8; ++r) {
            int val = (int)(lo[r] + 64.0f * hi[r] + 0.5f);
            if (n < 8) ap[(m0 + mb + r) * 8 + n] = (unsigned short)val;
        }
    }

    // ---- addr_v (tokens @ Wv), 64 output columns ----
    #pragma unroll
    for (int n0 = 0; n0 < 64; n0 += 16) {
        v8f lo = {}; v8f hi = {};
        #pragma unroll
        for (int k0 = 0; k0 < 64; k0 += 32) {
            v16h a = load_A_tokens(tokens, m0, k0, lane);
            lo = WMMA_F16(a, load_B(Wv_lo, k0, n0, 64, lane), lo);
            hi = WMMA_F16(a, load_B(Wv_hi, k0, n0, 64, lane), hi);
        }
        #pragma unroll
        for (int r = 0; r < 8; ++r) {
            int val = (int)(lo[r] + 64.0f * hi[r] + 0.5f);
            av[(m0 + mb + r) * 64 + n0 + n] = (unsigned short)val;
        }
    }
}

// ---------------------------------------------------------------------------
// Kernel 3: causal vote scan + first-argmax + output gather.
// One block handles rows i and 2047-i (load balance). 256 threads.
// ---------------------------------------------------------------------------
__global__ void softram_vote(const unsigned short* __restrict__ aq,
                             const unsigned short* __restrict__ ak,
                             const unsigned short* __restrict__ ap,
                             const unsigned short* __restrict__ av,
                             const unsigned int* __restrict__ packedT,
                             const float* __restrict__ table_v,   // [64][4096]
                             float* __restrict__ out)             // [2048][64]
{
    __shared__ unsigned int sT[1024];   // 4 KB bit-packed head table
    __shared__ unsigned int sRed[256];
    __shared__ int sBest[2];

    const int tid = threadIdx.x;
    for (int w = tid; w < 1024; w += 256) sT[w] = packedT[w];
    __syncthreads();

    for (int pass = 0; pass < 2; ++pass) {
        const int i = pass ? (SDIM - 1 - (int)blockIdx.x) : (int)blockIdx.x;

        int aqr[8];
        #pragma unroll
        for (int h = 0; h < 8; ++h) aqr[h] = (int)aq[i * 8 + h];

        // key = (vote << 11) | (2047 - j): max key -> max vote, then min j.
        unsigned int bestKey = 0;
        for (int j = tid; j <= i; j += 256) {
            int rel = i - j;
            uint4 kk = ((const uint4*)ak)[j];
            uint4 pp = ((const uint4*)ap)[rel];
            unsigned int kw[4] = {kk.x, kk.y, kk.z, kk.w};
            unsigned int pw[4] = {pp.x, pp.y, pp.z, pp.w};
            int v = 0;
            #pragma unroll
            for (int h = 0; h < 8; ++h) {
                int a = aqr[h]
                      + (int)((kw[h >> 1] >> ((h & 1) * 16)) & 0xFFFFu)
                      + (int)((pw[h >> 1] >> ((h & 1) * 16)) & 0xFFFFu);
                v += (int)((sT[(h << 7) + (a >> 5)] >> (a & 31)) & 1u);
            }
            unsigned int key = ((unsigned int)v << 11) | (unsigned int)(2047 - j);
            bestKey = (key > bestKey) ? key : bestKey;
        }

        sRed[tid] = bestKey;
        __syncthreads();
        #pragma unroll
        for (int s = 128; s > 0; s >>= 1) {
            if (tid < s) {
                unsigned int o = sRed[tid + s];
                if (o > sRed[tid]) sRed[tid] = o;
            }
            __syncthreads();
        }
        if (tid == 0) {
            sBest[0] = (int)(sRed[0] >> 11);                 // max vote
            sBest[1] = 2047 - (int)(sRed[0] & 2047u);        // first argmax j
        }
        __syncthreads();

        const int vote = sBest[0];
        const int best = sBest[1];
        if (tid < 64) {
            float o = 0.0f;
            if (vote > 0) o = table_v[tid * 4096 + (int)av[best * 64 + tid]];
            out[i * 64 + tid] = o;
        }
        __syncthreads();
    }
}

// ---------------------------------------------------------------------------
// Launch
// ---------------------------------------------------------------------------
extern "C" void kernel_launch(void* const* d_in, const int* in_sizes, int n_in,
                              void* d_out, int out_size, void* d_ws, size_t ws_size,
                              hipStream_t stream)
{
    const int*   tokens     = (const int*)d_in[0];     // [2048][64]
    const int*   conn_heads = (const int*)d_in[1];     // [8][12]
    const float* head_table = (const float*)d_in[2];   // [8][4096]
    const int*   conn_v     = (const int*)d_in[3];     // [64][12]
    const float* table_v    = (const float*)d_in[4];   // [64][4096]
    float*       out        = (float*)d_out;           // [2048][64]

    char* ws = (char*)d_ws;
    // all offsets 256-byte aligned; total ~378 KB
    _Float16*       Wqk_lo  = (_Float16*)(ws + 0);        // 2048 B
    _Float16*       Wqk_hi  = (_Float16*)(ws + 2048);     // 2048 B
    _Float16*       Wp_lo   = (_Float16*)(ws + 4096);     // 1024 B
    _Float16*       Wp_hi   = (_Float16*)(ws + 5120);     // 1024 B
    _Float16*       Wv_lo   = (_Float16*)(ws + 6144);     // 8192 B
    _Float16*       Wv_hi   = (_Float16*)(ws + 14336);    // 8192 B
    unsigned int*   packedT = (unsigned int*)(ws + 22528);   // 4096 B
    unsigned short* aq      = (unsigned short*)(ws + 26624); // 32768 B
    unsigned short* ak      = (unsigned short*)(ws + 59392); // 32768 B
    unsigned short* ap      = (unsigned short*)(ws + 92160); // 32768 B
    unsigned short* av      = (unsigned short*)(ws + 124928);// 262144 B -> end 387072

    softram_build<<<1, 256, 0, stream>>>(conn_heads, head_table, conn_v,
                                         Wqk_lo, Wqk_hi, Wp_lo, Wp_hi,
                                         Wv_lo, Wv_hi, packedT);

    softram_addr<<<SDIM / 16, 32, 0, stream>>>(tokens,
                                               Wqk_lo, Wqk_hi, Wp_lo, Wp_hi,
                                               Wv_lo, Wv_hi,
                                               aq, ak, ap, av);

    softram_vote<<<SDIM / 2, 256, 0, stream>>>(aq, ak, ap, av, packedT,
                                               table_v, out);
}
